// BinaryLinear_24790551232953
// MI455X (gfx1250) — matrix operand order repaired and checked by reading, not compiled, not run on hardware
//
#include <hip/hip_runtime.h>
#include <cstdint>

// BinaryLinear on MI455X (gfx1250, wave32, WMMA):
//   out[t,o] = (sum_k x[t,k] * sign(w[o,k])) * scale[o] + bias[o]
// scale[o] = max(mean|w[o,:]|, 1e-6) factors out of the K reduction, so the
// GEMM runs in bf16 (weights are exactly +-1 in bf16) with fp32 accumulation,
// and scale/bias are applied in the fp32 epilogue.

#define TOKENS 8192
#define INF    4096
#define OUTF   4096

#define BM 128
#define BN 128
#define BK 32
#define LDSS 40   // padded LDS row stride in bf16 elements (80B) -> conflict-free b128

typedef __attribute__((ext_vector_type(16))) __bf16 v16bf;
typedef __attribute__((ext_vector_type(8)))  float  v8f;
typedef __attribute__((ext_vector_type(4)))  int    v4i;

union FragU { uint4 u[2]; v16bf v; };   // 32B: one WMMA A/B fragment per lane
union AccU  { v8f v; float f[8]; };

// ---- CDNA5 async global->LDS copy (ASYNCcnt DMA), guarded for portability --
#if __has_builtin(__builtin_amdgcn_global_load_async_to_lds_b128)
#define USE_ASYNC_LDS 1
typedef __attribute__((address_space(1))) v4i glb_v4i;   // global int4 (64-bit ptr)
typedef __attribute__((address_space(3))) v4i lds_v4i;   // LDS int4 (32-bit ptr)
static __device__ __forceinline__ void async_cp16(const void* gsrc, const void* ldst) {
  // low 32 bits of a flat pointer into the LDS aperture == LDS byte address
  glb_v4i* s = (glb_v4i*)(uintptr_t)gsrc;
  lds_v4i* d = (lds_v4i*)(unsigned)(uintptr_t)ldst;
  __builtin_amdgcn_global_load_async_to_lds_b128(s, d, 0, 0);
}
static __device__ __forceinline__ void async_wait0() {
#if __has_builtin(__builtin_amdgcn_s_wait_asynccnt)
  __builtin_amdgcn_s_wait_asynccnt(0);
#else
  asm volatile("s_wait_asynccnt 0x0" ::: "memory");
#endif
}
#else
#define USE_ASYNC_LDS 0
#endif

static __device__ __forceinline__ unsigned f2bf(float f) {
  // round-to-nearest-even fp32 -> bf16
  unsigned u = __float_as_uint(f);
  return ((u + 0x7FFFu + ((u >> 16) & 1u)) >> 16) & 0xFFFFu;
}

// ---------------------------------------------------------------------------
// Kernel 1: per-row scale = max(mean|w|, 1e-6); pack sign(w) as bf16 +-1.
// One 256-thread block per output row (4096 fp32 elements).
// ---------------------------------------------------------------------------
__global__ __launch_bounds__(256)
void binarize_weight(const float* __restrict__ w,
                     unsigned short* __restrict__ wb,
                     float* __restrict__ scale) {
  __shared__ float red[256];
  const int row = blockIdx.x;
  const int t   = threadIdx.x;
  const float4* wr  = reinterpret_cast<const float4*>(w + (size_t)row * INF);
  uint2*        wbr = reinterpret_cast<uint2*>(wb + (size_t)row * INF);

  float s = 0.f;
#pragma unroll
  for (int i = 0; i < 4; ++i) {
    float4 v = wr[t + i * 256];
    s += fabsf(v.x) + fabsf(v.y) + fabsf(v.z) + fabsf(v.w);
    unsigned b0 = 0x3F80u | ((__float_as_uint(v.x) >> 31) << 15); // +-1.0 bf16
    unsigned b1 = 0x3F80u | ((__float_as_uint(v.y) >> 31) << 15);
    unsigned b2 = 0x3F80u | ((__float_as_uint(v.z) >> 31) << 15);
    unsigned b3 = 0x3F80u | ((__float_as_uint(v.w) >> 31) << 15);
    uint2 q; q.x = b0 | (b1 << 16); q.y = b2 | (b3 << 16);
    wbr[t + i * 256] = q;
  }
  red[t] = s;
  __syncthreads();
#pragma unroll
  for (int off = 128; off > 0; off >>= 1) {
    if (t < off) red[t] += red[t + off];
    __syncthreads();
  }
  if (t == 0) scale[row] = fmaxf(red[0] * (1.0f / (float)INF), 1e-6f);
}

// ---------------------------------------------------------------------------
// Kernel 2: fp32 -> bf16 (RNE) for the activations.
// ---------------------------------------------------------------------------
__global__ __launch_bounds__(256)
void cvt_x(const float4* __restrict__ x, uint2* __restrict__ xb, int n4) {
  int idx    = blockIdx.x * blockDim.x + threadIdx.x;
  int stride = gridDim.x * blockDim.x;
  for (int i = idx; i < n4; i += stride) {
    float4 v = x[i];
    uint2 q;
    q.x = f2bf(v.x) | (f2bf(v.y) << 16);
    q.y = f2bf(v.z) | (f2bf(v.w) << 16);
    xb[i] = q;
  }
}

// ---------------------------------------------------------------------------
// Kernel 3: bf16 WMMA GEMM. 128x128 block tile, BK=32, 256 threads = 8 waves
// arranged 2(M) x 4(N); each wave owns 64x32 = 4x2 v_wmma_f32_16x16x32_bf16.
// Double-buffered LDS; tile k+1 is DMA'd (async-to-LDS) while tile k computes.
// ---------------------------------------------------------------------------
__global__ __launch_bounds__(256)
void gemm_bf16_wmma(const unsigned short* __restrict__ Xb,   // [TOKENS][INF] bf16
                    const unsigned short* __restrict__ Wb,   // [OUTF][INF]  bf16 (+-1)
                    const float* __restrict__ scale,         // [OUTF]
                    const float* __restrict__ bias,          // [OUTF]
                    float* __restrict__ out) {               // [TOKENS][OUTF]
  __shared__ __align__(16) unsigned short lA[2][BM * LDSS];
  __shared__ __align__(16) unsigned short lB[2][BN * LDSS];

  const int tid   = threadIdx.x;
  const int lane  = tid & 31;
  const int lm    = lane & 15;   // M row (A frag) / N col (B frag) within 16x16
  const int khalf = lane >> 4;   // which K half-group this lane holds
  const int wid   = tid >> 5;
  const int waveM = wid >> 2;    // 0..1 -> 64-row M sub-band
  const int waveN = wid & 3;     // 0..3 -> 32-col N sub-band

  const int mBlock = blockIdx.y * BM;
  const int nBlock = blockIdx.x * BN;

  // Tile fill: 512 16B chunks per tile, 2 per thread (8 bf16 each).
  const int r0 = tid >> 2;          const int kc0 = tid & 3;   // rows 0..63
  const int r1 = (tid + 256) >> 2;  const int kc1 = tid & 3;   // rows 64..127

  const uint4* gA = reinterpret_cast<const uint4*>(Xb);
  const uint4* gB = reinterpret_cast<const uint4*>(Wb);
  const size_t rowQ = INF / 8;   // uint4 per global row (=512)

  const uint4* gA0 = gA + (size_t)(mBlock + r0) * rowQ + kc0;
  const uint4* gA1 = gA + (size_t)(mBlock + r1) * rowQ + kc1;
  const uint4* gB0 = gB + (size_t)(nBlock + r0) * rowQ + kc0;
  const uint4* gB1 = gB + (size_t)(nBlock + r1) * rowQ + kc1;

  const int eA0 = r0 * LDSS + kc0 * 8;
  const int eA1 = r1 * LDSS + kc1 * 8;

  // ---- preload K-tile 0 into LDS buffer 0 ----
#if USE_ASYNC_LDS
  async_cp16(gA0, &lA[0][eA0]);
  async_cp16(gA1, &lA[0][eA1]);
  async_cp16(gB0, &lB[0][eA0]);
  async_cp16(gB1, &lB[0][eA1]);
  async_wait0();
#else
  {
    uint4 a0 = *gA0, a1 = *gA1, b0 = *gB0, b1 = *gB1;
    *reinterpret_cast<uint4*>(&lA[0][eA0]) = a0;
    *reinterpret_cast<uint4*>(&lA[0][eA1]) = a1;
    *reinterpret_cast<uint4*>(&lB[0][eA0]) = b0;
    *reinterpret_cast<uint4*>(&lB[0][eA1]) = b1;
  }
#endif
  __syncthreads();

  const v8f vzero = {0.f, 0.f, 0.f, 0.f, 0.f, 0.f, 0.f, 0.f};
  v8f acc[4][2];
#pragma unroll
  for (int i = 0; i < 4; ++i)
#pragma unroll
    for (int j = 0; j < 2; ++j) acc[i][j] = vzero;

  const int KT = INF / BK;       // 128 K-steps
  for (int kt = 0; kt < KT; ++kt) {
    const int cur = kt & 1, nxt = cur ^ 1;
    const bool more = (kt + 1) < KT;
    const size_t ko = (size_t)(kt + 1) * (BK / 8);

#if USE_ASYNC_LDS
    // DMA next tile directly into the other LDS buffer (no VGPR staging)
    if (more) {
      async_cp16(gA0 + ko, &lA[nxt][eA0]);
      async_cp16(gA1 + ko, &lA[nxt][eA1]);
      async_cp16(gB0 + ko, &lB[nxt][eA0]);
      async_cp16(gB1 + ko, &lB[nxt][eA1]);
    }
#else
    uint4 na0, na1, nb0, nb1;
    if (more) {
      na0 = gA0[ko]; na1 = gA1[ko]; nb0 = gB0[ko]; nb1 = gB1[ko];
    }
#endif

    // --- load WMMA fragments from LDS (ISA 7.12.2 16-bit layouts) ---
    // A 16x32: lane holds M=lm, K = khalf*8+0..7 and 16+khalf*8+0..7
    FragU af[4];
#pragma unroll
    for (int i = 0; i < 4; ++i) {
      const int row = waveM * 64 + i * 16 + lm;
      const int e0  = row * LDSS + khalf * 8;
      af[i].u[0] = *reinterpret_cast<const uint4*>(&lA[cur][e0]);
      af[i].u[1] = *reinterpret_cast<const uint4*>(&lA[cur][e0 + 16]);
    }
    // B 32x16: lane holds N=lm, K = khalf*8+0..7 and 16+khalf*8+0..7
    FragU bf[2];
#pragma unroll
    for (int j = 0; j < 2; ++j) {
      const int row = waveN * 32 + j * 16 + lm;
      const int e0  = row * LDSS + khalf * 8;
      bf[j].u[0] = *reinterpret_cast<const uint4*>(&lB[cur][e0]);
      bf[j].u[1] = *reinterpret_cast<const uint4*>(&lB[cur][e0 + 16]);
    }

    // --- 8 WMMAs: D = A x B + C, fp32 accumulate ---
#pragma unroll
    for (int i = 0; i < 4; ++i)
#pragma unroll
      for (int j = 0; j < 2; ++j)
        acc[i][j] = __builtin_amdgcn_wmma_f32_16x16x32_bf16(
            /*neg_a=*/false, af[i].v, /*neg_b=*/false, bf[j].v,
            /*c_mod=*/(short)0, acc[i][j],
            /*reuse_a=*/false, /*reuse_b=*/false);

#if USE_ASYNC_LDS
    async_wait0();         // next tile landed in LDS before the barrier
#else
    if (more) {
      *reinterpret_cast<uint4*>(&lA[nxt][eA0]) = na0;
      *reinterpret_cast<uint4*>(&lA[nxt][eA1]) = na1;
      *reinterpret_cast<uint4*>(&lB[nxt][eA0]) = nb0;
      *reinterpret_cast<uint4*>(&lB[nxt][eA1]) = nb1;
    }
#endif
    __syncthreads();
  }

  // --- epilogue: apply per-output-column scale + bias, write fp32 ---
  // C/D layout: lane -> column N=lm; VGPR r -> row M = khalf*8 + r.
#pragma unroll
  for (int j = 0; j < 2; ++j) {
    const int n  = nBlock + waveN * 32 + j * 16 + lm;
    const float sc = scale[n];
    const float bs = bias[n];
#pragma unroll
    for (int i = 0; i < 4; ++i) {
      const int m0 = mBlock + waveM * 64 + i * 16 + khalf * 8;
      AccU u; u.v = acc[i][j];
      float* op = out + (size_t)m0 * OUTF + n;
#pragma unroll
      for (int r = 0; r < 8; ++r)
        op[(size_t)r * OUTF] = u.f[r] * sc + bs;
    }
  }
}

// ---------------------------------------------------------------------------
// Launch: d_in = {inputs[8192x4096] f32, weight[4096x4096] f32, bias[4096] f32}
// d_ws layout: [scale 16KB][W_bf16 32MB][X_bf16 64MB]  (~96MB scratch)
// ---------------------------------------------------------------------------
extern "C" void kernel_launch(void* const* d_in, const int* in_sizes, int n_in,
                              void* d_out, int out_size, void* d_ws, size_t ws_size,
                              hipStream_t stream) {
  const float* x    = (const float*)d_in[0];
  const float* w    = (const float*)d_in[1];
  const float* bias = (const float*)d_in[2];
  float* out = (float*)d_out;

  char* ws = (char*)d_ws;
  float*          scale = (float*)ws;                                   // 16 KB
  unsigned short* wb    = (unsigned short*)(ws + 16384);                // 32 MB
  unsigned short* xb    = (unsigned short*)(ws + 16384 +
                                            (size_t)OUTF * INF * 2);    // 64 MB

  binarize_weight<<<OUTF, 256, 0, stream>>>(w, wb, scale);
  cvt_x<<<8192, 256, 0, stream>>>((const float4*)x, (uint2*)xb,
                                  TOKENS * INF / 4);
  dim3 grid(OUTF / BN, TOKENS / BM);   // x-major: N tiles fastest -> A-band L2 reuse
  gemm_bf16_wmma<<<grid, 256, 0, stream>>>(xb, wb, scale, bias, out);
}
